// Spectre_38517266710667
// MI455X (gfx1250) — matrix operand order repaired and checked by reading, not compiled, not run on hardware
//
#include <hip/hip_runtime.h>
#include <hip/hip_bf16.h>

typedef __attribute__((ext_vector_type(16))) _Float16 v16h;
typedef __attribute__((ext_vector_type(8)))  _Float16 h8;
typedef __attribute__((ext_vector_type(8)))  float    v8f;

#define LDS_STRIDE 40   // halves; 80B row pitch, 16B aligned, bank-spread
#define HS_STRIDE  2080 // halves per shifted filter copy; 4160B = 16B multiple

// ---------- workspace layout (bytes) ----------
static constexpr size_t OFF_XH   = 0;                         // f16 8192x1024   = 16 MB
static constexpr size_t OFF_WVH  = OFF_XH   + 16777216;       // f16 1024x1024   =  2 MB
static constexpr size_t OFF_WOH  = OFF_WVH  + 2097152;        // f16 1024x1024   =  2 MB
static constexpr size_t OFF_VHT  = OFF_WOH  + 2097152;        // f16 (b,h,d,n)   = 16 MB
static constexpr size_t OFF_VTH  = OFF_VHT  + 16777216;       // f16 (b,n,h*d)   = 16 MB
static constexpr size_t OFF_XBAR = OFF_VTH  + 16777216;       // f32 4x1024
static constexpr size_t OFF_SG   = OFF_XBAR + 16384;          // f32 64x2048
static constexpr size_t OFF_HF   = OFF_SG   + 524288;         // f16 64x2048

__device__ __forceinline__ v16h make_v16(h8 lo, h8 hi) {
  return __builtin_shufflevector(lo, hi, 0,1,2,3,4,5,6,7,8,9,10,11,12,13,14,15);
}

// A fragment: 16 rows x 32 K, row-major LDS tile (stride LDS_STRIDE halves).
// ISA layout: lanes 0-15 K=0..7,16..23 ; lanes 16-31 K=8..15,24..31 ; M = lane%16.
__device__ __forceinline__ v16h load_frag_A(const _Float16* lds, int lane) {
  int row = lane & 15;
  int kb  = (lane >> 4) << 3;               // 0 or 8
  const _Float16* p = lds + row * LDS_STRIDE;
  h8 lo = *(const h8*)(p + kb);
  h8 hi = *(const h8*)(p + 16 + kb);
  return make_v16(lo, hi);
}

// B fragment: ldsB[n][k] tile; lane holds column n=lane%16
// (lanes 0-15: K 0..7,16..23, lanes 16-31: K 8..15,24..31).
__device__ __forceinline__ v16h load_frag_B(const _Float16* ldsB, int nf, int lane) {
  int col = nf + (lane & 15);
  int kb  = (lane >> 4) << 3;               // 0 or 8
  const _Float16* p = ldsB + col * LDS_STRIDE + kb;
  h8 lo = *(const h8*)(p);
  h8 hi = *(const h8*)(p + 16);
  return make_v16(lo, hi);
}

// ---------------- prep kernels ----------------

__global__ void cast_w_kernel(const float* __restrict__ wv, const float* __restrict__ wo,
                              _Float16* __restrict__ wvh, _Float16* __restrict__ woh) {
  int i = blockIdx.x * blockDim.x + threadIdx.x;
  wvh[i] = (_Float16)wv[i];
  woh[i] = (_Float16)wo[i];
}

// column means of x (mean over N) + cast x -> f16 in one pass
__global__ void reduce_cast_x(const float* __restrict__ x, _Float16* __restrict__ xh,
                              float* __restrict__ xbar) {
  int idx = blockIdx.x * blockDim.x + threadIdx.x;      // 0..4095 : (b,d)
  int b = idx >> 10, d = idx & 1023;
  const float* xp  = x  + (size_t)b * 2048 * 1024 + d;
  _Float16*    xhp = xh + (size_t)b * 2048 * 1024 + d;
  float s = 0.f;
  for (int n = 0; n < 2048; ++n) {
    float v = xp[(size_t)n * 1024];
    s += v;
    xhp[(size_t)n * 1024] = (_Float16)v;
  }
  xbar[idx] = s * (1.0f / 2048.0f);
}

// per (b,h): bar (LN) -> g -> hfilt = irfft(gc) ; sg gate. One block each.
__global__ void head_small(const float* __restrict__ xbar, const float* __restrict__ w_q,
                           const float* __restrict__ ln_w, const float* __restrict__ ln_b,
                           const float* __restrict__ g_up, const float* __restrict__ g_dnF,
                           const float* __restrict__ wu,   const float* __restrict__ wd,
                           float* __restrict__ sg, _Float16* __restrict__ hfilt) {
  __shared__ float barS[64];
  __shared__ float uS[256], suS[256];
  __shared__ float gS[2050];
  __shared__ float twc[2048], tws[2048];
  const int t = threadIdx.x;
  const int bh = blockIdx.x, b = bh >> 4, h = bh & 15;
  const float* xb = xbar + b * 1024;

  for (int p = t; p < 2048; p += 256) {
    float s, c;
    __sincosf(6.28318530717958647692f * (float)p * (1.0f / 2048.0f), &s, &c);
    twc[p] = c; tws[p] = s;
  }
  if (t < 64) {                                   // bar = xbar @ w_q.T (head slice)
    const float* wr = w_q + (size_t)(h * 64 + t) * 1024;
    float s = 0.f;
    for (int k = 0; k < 1024; ++k) s += xb[k] * wr[k];
    barS[t] = s;
  }
  __syncthreads();
  if (t == 0) {                                   // layernorm over dh=64
    float mu = 0.f;
    for (int d = 0; d < 64; ++d) mu += barS[d];
    mu *= (1.0f / 64.0f);
    float var = 0.f;
    for (int d = 0; d < 64; ++d) { float e = barS[d] - mu; var += e * e; }
    var *= (1.0f / 64.0f);
    float rs = rsqrtf(var + 1e-6f);
    for (int d = 0; d < 64; ++d) barS[d] = (barS[d] - mu) * rs * ln_w[d] + ln_b[d];
  }
  __syncthreads();
  {                                               // u = silu(bar@g_up.T), su = silu(bar@wu.T)
    float s1 = 0.f, s2 = 0.f;
    const float* g1 = g_up + (size_t)t * 64;
    const float* g2 = wu   + (size_t)t * 64;
    for (int d = 0; d < 64; ++d) { s1 += barS[d] * g1[d]; s2 += barS[d] * g2[d]; }
    uS[t]  = s1 / (1.f + __expf(-s1));
    suS[t] = s2 / (1.f + __expf(-s2));
  }
  __syncthreads();
  for (int i = t; i < 2050; i += 256) {           // g = u @ g_dnF.T
    const float* gr = g_dnF + (size_t)i * 256;
    float s = 0.f;
    for (int j = 0; j < 256; ++j) s += uS[j] * gr[j];
    gS[i] = s;
  }
  for (int n = t; n < 2048; n += 256) {           // sg = su @ wd.T
    const float* wr = wd + (size_t)n * 256;
    float s = 0.f;
    for (int j = 0; j < 256; ++j) s += suS[j] * wr[j];
    sg[(size_t)bh * 2048 + n] = s;
  }
  __syncthreads();
  // hfilt[p] = (1/N)[gr0 + (-1)^p * gr_{N/2} + 2*sum_{k=1}^{N/2-1}(gr_k cos - gi_k sin)]
  for (int p = t; p < 2048; p += 256) {
    float k0t  = gS[0];
    float klt  = gS[2048] * ((p & 1) ? -1.f : 1.f);
    float acc  = 0.f;
    int idx = p;
    for (int k = 1; k < 1024; ++k) {
      acc += gS[2 * k] * twc[idx] - gS[2 * k + 1] * tws[idx];
      idx = (idx + p) & 2047;
    }
    float hv = (k0t + klt + 2.f * acc) * (1.0f / 2048.0f);
    hfilt[(size_t)bh * 2048 + p] = (_Float16)hv;
  }
}

// ---------------- WMMA GEMMs (double-buffered LDS) ----------------
// C(8192x1024) = A(8192x1024,f16) @ Bw^T (Bw row-major 1024x1024,f16)
// EPI 0: vector-store f16 into transposed v layout (b,h,d,n).
// EPI 1: f32 row-major to out.
template <int EPI>
__global__ void gemm16(const _Float16* __restrict__ A, const _Float16* __restrict__ Bw,
                       _Float16* __restrict__ outH, float* __restrict__ outF) {
  __shared__ __align__(16) _Float16 ldsA[2][64 * LDS_STRIDE];
  __shared__ __align__(16) _Float16 ldsB[2][64 * LDS_STRIDE];
  const int t = threadIdx.x, wave = t >> 5, lane = t & 31;
  const int m0 = blockIdx.x * 64, n0 = blockIdx.y * 64;
  const int srow = t >> 2, spart = t & 3;           // staging: rows srow, srow+32
  const _Float16* gA = A  + (size_t)m0 * 1024;
  const _Float16* gB = Bw + (size_t)n0 * 1024;

#pragma unroll
  for (int it = 0; it < 2; ++it) {                  // prologue: stage tile k0=0
    int row = srow + it * 32;
    *(uint4*)(&ldsA[0][row * LDS_STRIDE + spart * 8]) =
        *(const uint4*)(gA + (size_t)row * 1024 + spart * 8);
    *(uint4*)(&ldsB[0][row * LDS_STRIDE + spart * 8]) =
        *(const uint4*)(gB + (size_t)row * 1024 + spart * 8);
  }
  __syncthreads();

  v8f acc[4] = {};
  int buf = 0;
  for (int k0 = 0; k0 < 1024; k0 += 32, buf ^= 1) {
    uint4 ra[2], rb[2];
    const bool more = (k0 + 32) < 1024;
    if (more) {                                     // prefetch next tile into regs
#pragma unroll
      for (int it = 0; it < 2; ++it) {
        int row = srow + it * 32;
        ra[it] = *(const uint4*)(gA + (size_t)row * 1024 + (k0 + 32) + spart * 8);
        rb[it] = *(const uint4*)(gB + (size_t)row * 1024 + (k0 + 32) + spart * 8);
      }
    }
    v16h af = load_frag_A(&ldsA[buf][wave * 16 * LDS_STRIDE], lane);
#pragma unroll
    for (int tb = 0; tb < 4; ++tb) {
      v16h bf = load_frag_B(&ldsB[buf][0], tb * 16, lane);
      acc[tb] = __builtin_amdgcn_wmma_f32_16x16x32_f16(false, af, false, bf,
                                                       (short)0, acc[tb], false, false);
    }
    if (more) {                                     // commit next tile to other buffer
#pragma unroll
      for (int it = 0; it < 2; ++it) {
        int row = srow + it * 32;
        *(uint4*)(&ldsA[buf ^ 1][row * LDS_STRIDE + spart * 8]) = ra[it];
        *(uint4*)(&ldsB[buf ^ 1][row * LDS_STRIDE + spart * 8]) = rb[it];
      }
    }
    __syncthreads();
  }

  const int col16 = lane & 15, row0 = (lane >> 4) << 3;
  const int mb = m0 + wave * 16 + row0;             // 8 consecutive m rows
#pragma unroll
  for (int tb = 0; tb < 4; ++tb) {
    int n = n0 + tb * 16 + col16;
    if (EPI == 0) {
      // v transposed: (b, h, d, n) ; 8 consecutive seq positions -> one h8 store
      int b = mb >> 11, nseq = mb & 2047, hh = n >> 6, d = n & 63;
      h8 pk;
#pragma unroll
      for (int r = 0; r < 8; ++r) pk[r] = (_Float16)acc[tb][r];
      *(h8*)(outH + ((size_t)(b * 16 + hh) * 64 + d) * 2048 + nseq) = pk;
    } else {
#pragma unroll
      for (int r = 0; r < 8; ++r)
        outF[(size_t)(mb + r) * 1024 + n] = acc[tb][r];
    }
  }
}

// vt(2048x64) = Circ(hfilt)(2048x2048) @ v_head(2048x64), per (b,h).
// vhT is (b,h,d,n) so B tiles stage with aligned uint4 copies.
// Circulant A frags come from 8 shift-replicated reversed-filter copies in LDS,
// so each frag is two aligned ds_load_b128.  dwt/idwt gate fused in epilogue.
__global__ void conv_gemm(const _Float16* __restrict__ vhT, const _Float16* __restrict__ hfilt,
                          const float* __restrict__ sg, _Float16* __restrict__ vth) {
  __shared__ __align__(16) _Float16 ldsHf[2048];
  __shared__ __align__(16) _Float16 HS[8 * HS_STRIDE];   // HS[s][i] = h[-(i+s) mod 2048]
  __shared__ __align__(16) _Float16 ldsB[2][64 * LDS_STRIDE];
  const int t = threadIdx.x, wave = t >> 5, lane = t & 31;
  const int bh = blockIdx.x, m0 = blockIdx.y * 64;
  const _Float16* vhead = vhT + (size_t)bh * 64 * 2048;   // [d][n]
  const int srow = t >> 2, spart = t & 3;
  {
    const uint4* src = (const uint4*)(hfilt + (size_t)bh * 2048);
    uint4* dst = (uint4*)ldsHf;
    dst[t * 2]     = src[t * 2];
    dst[t * 2 + 1] = src[t * 2 + 1];
  }
#pragma unroll
  for (int it = 0; it < 2; ++it) {                  // prologue: stage B tile k0=0
    int d = srow + it * 32;
    *(uint4*)(&ldsB[0][d * LDS_STRIDE + spart * 8]) =
        *(const uint4*)(vhead + (size_t)d * 2048 + spart * 8);
  }
  __syncthreads();
  for (int s = 0; s < 8; ++s)                       // build shifted reversed copies
    for (int i = t; i < HS_STRIDE; i += 128)
      HS[s * HS_STRIDE + i] = ldsHf[(4096 - i - s) & 2047];
  __syncthreads();

  v8f acc[4] = {};
  const int mrow = m0 + wave * 16 + (lane & 15);
  const int kb2  = (lane >> 4) << 3;
  const _Float16* HSrow = HS + ((-mrow) & 7) * HS_STRIDE;  // shift constant over K loop
  int buf = 0;
  for (int k0 = 0; k0 < 2048; k0 += 32, buf ^= 1) {
    uint4 rb[2];
    const bool more = (k0 + 32) < 2048;
    if (more) {
#pragma unroll
      for (int it = 0; it < 2; ++it) {
        int d = srow + it * 32;
        rb[it] = *(const uint4*)(vhead + (size_t)d * 2048 + (k0 + 32) + spart * 8);
      }
    }
    int bidx = (k0 + kb2 - mrow) & 2047;            // gather base in reversed filter
    int qo = (bidx >> 3) << 3;                      // aligned element offset
    h8 alo = *(const h8*)(HSrow + qo);
    h8 ahi = *(const h8*)(HSrow + qo + 16);
    v16h af = make_v16(alo, ahi);
#pragma unroll
    for (int tb = 0; tb < 4; ++tb) {
      v16h bf = load_frag_B(&ldsB[buf][0], tb * 16, lane);
      acc[tb] = __builtin_amdgcn_wmma_f32_16x16x32_f16(false, af, false, bf,
                                                       (short)0, acc[tb], false, false);
    }
    if (more) {
#pragma unroll
      for (int it = 0; it < 2; ++it) {
        int d = srow + it * 32;
        *(uint4*)(&ldsB[buf ^ 1][d * LDS_STRIDE + spart * 8]) = rb[it];
      }
    }
    __syncthreads();
  }

  // fused dwt -> gate -> idwt residual on 4 in-register row pairs, write f16 (b,n,H*d)
  const int col16 = lane & 15, row0 = (lane >> 4) << 3;
  const int mb = m0 + wave * 16 + row0;             // even, 8 consecutive seq rows
  const int b = bh >> 4, h = bh & 15;
  const int i0 = mb >> 1;
  float aC[4], bC[4];
#pragma unroll
  for (int rr = 0; rr < 4; ++rr) {
    float sgl = sg[(size_t)bh * 2048 + i0 + rr];
    float sgh = sg[(size_t)bh * 2048 + 1024 + i0 + rr];
    aC[rr] = 1.0f + 0.5f * (sgl + sgh);
    bC[rr] = 0.5f * (sgl - sgh);
  }
#pragma unroll
  for (int tb = 0; tb < 4; ++tb) {
    int d = tb * 16 + col16;
    size_t base = ((size_t)b * 2048 + mb) * 1024 + h * 64 + d;
#pragma unroll
    for (int rr = 0; rr < 4; ++rr) {
      float e = acc[tb][2 * rr], o = acc[tb][2 * rr + 1];
      vth[base + (size_t)(2 * rr) * 1024]     = (_Float16)(aC[rr] * e + bC[rr] * o);
      vth[base + (size_t)(2 * rr + 1) * 1024] = (_Float16)(bC[rr] * e + aC[rr] * o);
    }
  }
}

extern "C" void kernel_launch(void* const* d_in, const int* in_sizes, int n_in,
                              void* d_out, int out_size, void* d_ws, size_t ws_size,
                              hipStream_t stream) {
  const float* x     = (const float*)d_in[0];
  const float* w_q   = (const float*)d_in[1];
  const float* w_v   = (const float*)d_in[2];
  const float* w_o   = (const float*)d_in[3];
  const float* ln_w  = (const float*)d_in[4];
  const float* ln_b  = (const float*)d_in[5];
  const float* g_up  = (const float*)d_in[6];
  const float* g_dnF = (const float*)d_in[7];
  const float* wu    = (const float*)d_in[8];
  const float* wd    = (const float*)d_in[9];
  float* out = (float*)d_out;
  char* ws = (char*)d_ws;

  _Float16* xh    = (_Float16*)(ws + OFF_XH);
  _Float16* wvh   = (_Float16*)(ws + OFF_WVH);
  _Float16* woh   = (_Float16*)(ws + OFF_WOH);
  _Float16* vhT   = (_Float16*)(ws + OFF_VHT);
  _Float16* vth   = (_Float16*)(ws + OFF_VTH);
  float*    xbar  = (float*)   (ws + OFF_XBAR);
  float*    sg    = (float*)   (ws + OFF_SG);
  _Float16* hfilt = (_Float16*)(ws + OFF_HF);

  cast_w_kernel<<<4096, 256, 0, stream>>>(w_v, w_o, wvh, woh);
  reduce_cast_x<<<16, 256, 0, stream>>>(x, xh, xbar);
  head_small<<<64, 256, 0, stream>>>(xbar, w_q, ln_w, ln_b, g_up, g_dnF, wu, wd, sg, hfilt);
  gemm16<0><<<dim3(128, 16), 128, 0, stream>>>(xh, wvh, vhT, nullptr);     // v (transposed out)
  conv_gemm<<<dim3(64, 32), 128, 0, stream>>>(vhT, hfilt, sg, vth);        // spectral + wavelet
  gemm16<1><<<dim3(128, 16), 128, 0, stream>>>(vth, woh, nullptr, out);    // output proj
}
